// EGNNLayer_37177236914853
// MI455X (gfx1250) — matrix operand order repaired and checked by reading, not compiled, not run on hardware
//
#include <hip/hip_runtime.h>

// Problem constants (fixed by the reference).
#define NN 20000          // nodes
#define NE 320000         // edges
#define HD 256            // hidden dim
#define A_STR 520         // LDS stride (halfs) for K=512 staging: 1040B = 260 dwords = 4 mod 64 banks
#define M_STR 264         // LDS stride (halfs) for K=256 staging: 528B  = 132 dwords = 4 mod 64 banks
#define Y_STR 276         // LDS stride (floats) for LN staging
#define LN_EPS 1e-5f

typedef __attribute__((ext_vector_type(16))) _Float16 v16h;
typedef __attribute__((ext_vector_type(8)))  _Float16 v8h;
typedef __attribute__((ext_vector_type(8)))  float    v8f;

__device__ __forceinline__ float silu_f(float v) { return v / (1.0f + __expf(-v)); }

// A-fragment (16x32 f16, M=rows of the staged tile) from LDS, matching the
// CDNA5 16-bit A layout: lanes 0-15 -> K {kb..kb+7, kb+16..kb+23},
// lanes 16-31 -> K {kb+8..kb+15, kb+24..kb+31}. Two ds_load_b128 each.
__device__ __forceinline__ v16h afrag_lds(const _Float16* base, int stride, int kb, int lane) {
  const int m  = lane & 15;
  const int hi = lane >> 4;
  const _Float16* p = base + m * stride + kb + hi * 8;
  v8h lo = *(const v8h*)p;
  v8h hv = *(const v8h*)(p + 16);
  return __builtin_shufflevector(lo, hv, 0,1,2,3,4,5,6,7,8,9,10,11,12,13,14,15);
}

// B-fragment (32x16 f16) from a pre-transposed weight WT[N][K]:
// lane -> column n (lane&15), K {kb+h*8..+7, kb+16+h*8..+7} with h = lane>>4 (matches A).
__device__ __forceinline__ v16h bfrag_glb(const _Float16* __restrict__ wt, int K, int n, int kb, int lane) {
  const _Float16* p = wt + (size_t)n * K + kb + (lane >> 4) * 8;
  v8h lo = *(const v8h*)p;
  v8h hv = *(const v8h*)(p + 16);
  return __builtin_shufflevector(lo, hv, 0,1,2,3,4,5,6,7,8,9,10,11,12,13,14,15);
}

// ---------------- prep kernels ----------------
__global__ __launch_bounds__(256) void k_f32_to_f16(const float* __restrict__ src,
                                                    _Float16* __restrict__ dst, int n) {
  int i = blockIdx.x * 256 + threadIdx.x;
  if (i < n) dst[i] = (_Float16)src[i];
}

// wt[n*K + k] = (f16) w[k*N + n]   (w row-major [K][N])
__global__ __launch_bounds__(256) void k_transpose_f16(const float* __restrict__ w,
                                                       _Float16* __restrict__ wt, int K, int N) {
  int i = blockIdx.x * 256 + threadIdx.x;
  if (i < K * N) {
    int n = i / K, k = i - n * K;
    wt[i] = (_Float16)w[k * N + n];
  }
}

// ---------------- edge kernel: 16 edges per workgroup, 8 waves ----------------
__global__ __launch_bounds__(256) void egnn_edge_kernel(
    const _Float16* __restrict__ hF, const float* __restrict__ x,
    const int* __restrict__ eidx,                       // [2][E] (int32)
    const _Float16* __restrict__ eW1T,                  // [256][512] f16 (first 512 rows of e_w1, transposed)
    const float* __restrict__ eW1last,                  // e_w1 row 512 (dist_sq column), f32 [256]
    const float* __restrict__ eB1,
    const _Float16* __restrict__ eW2T, const float* __restrict__ eB2,
    const _Float16* __restrict__ cW1T, const float* __restrict__ cB1,
    const float* __restrict__ cW2,
    float* __restrict__ msg,                            // [N][256] f32, pre-zeroed (d_out h region)
    float* __restrict__ coordAgg)                       // [N][3]  f32, pre-zeroed (d_out x region)
{
  __shared__ __align__(16) _Float16 sA [16 * A_STR];    // edge_input f16 [16][512]
  __shared__ __align__(16) _Float16 sM [16 * M_STR];    // m         f16 [16][256]
  __shared__ __align__(16) _Float16 sM2[16 * M_STR];    // m_ij      f16 [16][256]
  __shared__ float sDiff[16][3];
  __shared__ float sDS[16];
  __shared__ float sCW[16];
  __shared__ int   sRow[16], sCol[16];

  const int tid  = threadIdx.x;
  const int lane = tid & 31;
  const int wave = tid >> 5;
  const int nc   = lane & 15;
  const int hi   = lane >> 4;
  const int na   = wave * 32 + nc;      // column for tile 2*wave
  const int nb   = na + 16;             // column for tile 2*wave+1
  const int eb   = blockIdx.x * 16;

  if (tid < 16) {
    int e = eb + tid;
    int r = eidx[e];
    int c = eidx[NE + e];
    sRow[tid] = r; sCol[tid] = c;
    float d0 = x[r * 3 + 0] - x[c * 3 + 0];
    float d1 = x[r * 3 + 1] - x[c * 3 + 1];
    float d2 = x[r * 3 + 2] - x[c * 3 + 2];
    sDiff[tid][0] = d0; sDiff[tid][1] = d1; sDiff[tid][2] = d2;
    sDS[tid] = d0 * d0 + d1 * d1 + d2 * d2;
    sCW[tid] = 0.0f;
  }
  __syncthreads();

  // Stage A = [h[row] | h[col]] as f16 (16 rows x 512), 16B chunks.
  for (int c = tid; c < 16 * 64; c += 256) {
    int m  = c >> 6;
    int k8 = (c & 63) * 8;
    const _Float16* src = (k8 < 256) ? (hF + (size_t)sRow[m] * HD + k8)
                                     : (hF + (size_t)sCol[m] * HD + (k8 - 256));
    *(v8h*)(sA + m * A_STR + k8) = *(const v8h*)src;
  }
  __syncthreads();

  // ---- GEMM1: [16x513] @ e_w1 -> silu -> sM ----
  {
    v8f c0 = {}, c1 = {};
#pragma unroll
    for (int ks = 0; ks < 16; ++ks) {
      v16h a  = afrag_lds(sA, A_STR, ks * 32, lane);
      v16h b0 = bfrag_glb(eW1T, 512, na, ks * 32, lane);
      v16h b1 = bfrag_glb(eW1T, 512, nb, ks * 32, lane);
      c0 = __builtin_amdgcn_wmma_f32_16x16x32_f16(false, a, false, b0, (short)0, c0, false, false);
      c1 = __builtin_amdgcn_wmma_f32_16x16x32_f16(false, a, false, b1, (short)0, c1, false, false);
    }
    float wlA = eW1last[na], wlB = eW1last[nb];
    float baA = eB1[na],     baB = eB1[nb];
#pragma unroll
    for (int r = 0; r < 8; ++r) {
      int m = r + 8 * hi;
      float vA = silu_f(c0[r] + sDS[m] * wlA + baA);
      float vB = silu_f(c1[r] + sDS[m] * wlB + baB);
      sM[m * M_STR + na] = (_Float16)vA;
      sM[m * M_STR + nb] = (_Float16)vB;
    }
  }
  __syncthreads();

  // ---- GEMM2: m @ e_w2 -> silu -> m_ij; scatter-add to msg; keep f16 copy ----
  {
    v8f c0 = {}, c1 = {};
#pragma unroll
    for (int ks = 0; ks < 8; ++ks) {
      v16h a  = afrag_lds(sM, M_STR, ks * 32, lane);
      v16h b0 = bfrag_glb(eW2T, 256, na, ks * 32, lane);
      v16h b1 = bfrag_glb(eW2T, 256, nb, ks * 32, lane);
      c0 = __builtin_amdgcn_wmma_f32_16x16x32_f16(false, a, false, b0, (short)0, c0, false, false);
      c1 = __builtin_amdgcn_wmma_f32_16x16x32_f16(false, a, false, b1, (short)0, c1, false, false);
    }
    float baA = eB2[na], baB = eB2[nb];
#pragma unroll
    for (int r = 0; r < 8; ++r) {
      int m = r + 8 * hi;
      float vA = silu_f(c0[r] + baA);
      float vB = silu_f(c1[r] + baB);
      sM2[m * M_STR + na] = (_Float16)vA;
      sM2[m * M_STR + nb] = (_Float16)vB;
      size_t rowb = (size_t)sCol[m] * HD;
      atomicAdd(&msg[rowb + na], vA);
      atomicAdd(&msg[rowb + nb], vB);
    }
  }
  __syncthreads();

  // ---- GEMM3: silu(m_ij @ c_w1 + b) @ c_w2 -> coord_weight[16] ----
  {
    v8f c0 = {}, c1 = {};
#pragma unroll
    for (int ks = 0; ks < 8; ++ks) {
      v16h a  = afrag_lds(sM2, M_STR, ks * 32, lane);
      v16h b0 = bfrag_glb(cW1T, 256, na, ks * 32, lane);
      v16h b1 = bfrag_glb(cW1T, 256, nb, ks * 32, lane);
      c0 = __builtin_amdgcn_wmma_f32_16x16x32_f16(false, a, false, b0, (short)0, c0, false, false);
      c1 = __builtin_amdgcn_wmma_f32_16x16x32_f16(false, a, false, b1, (short)0, c1, false, false);
    }
    float baA = cB1[na], baB = cB1[nb];
    float cwA = cW2[na], cwB = cW2[nb];
#pragma unroll
    for (int r = 0; r < 8; ++r) {
      int m = r + 8 * hi;
      float v = silu_f(c0[r] + baA) * cwA + silu_f(c1[r] + baB) * cwB;
      // reduce across the 16 columns held by this half-wave
      v += __shfl_xor(v, 8, 32);
      v += __shfl_xor(v, 4, 32);
      v += __shfl_xor(v, 2, 32);
      v += __shfl_xor(v, 1, 32);
      if (nc == 0) atomicAdd(&sCW[m], v);
    }
  }
  __syncthreads();

  if (tid < 16) {
    float cw = sCW[tid];
    size_t b = (size_t)sCol[tid] * 3;
    atomicAdd(&coordAgg[b + 0], cw * sDiff[tid][0]);
    atomicAdd(&coordAgg[b + 1], cw * sDiff[tid][1]);
    atomicAdd(&coordAgg[b + 2], cw * sDiff[tid][2]);
  }
}

// ---------------- node kernel: 16 nodes per workgroup ----------------
__global__ __launch_bounds__(256) void egnn_node_kernel(
    const float* __restrict__ h, const _Float16* __restrict__ hF,
    const float* __restrict__ msg,                      // aggregated messages (d_out h region)
    const _Float16* __restrict__ nW1T, const float* __restrict__ nB1,
    const _Float16* __restrict__ nW2T, const float* __restrict__ nB2,
    const float* __restrict__ lnG, const float* __restrict__ lnB,
    float* __restrict__ hOut)                           // same region as msg; staged first
{
  __shared__ __align__(16) _Float16 sA[16 * A_STR];     // [h | msg] f16 [16][512]
  __shared__ __align__(16) _Float16 sM[16 * M_STR];     // hidden    f16 [16][256]
  __shared__ float sY[16 * Y_STR];                      // y = h + h_new, f32

  const int tid  = threadIdx.x;
  const int lane = tid & 31;
  const int wave = tid >> 5;
  const int nc   = lane & 15;
  const int hi   = lane >> 4;
  const int na   = wave * 32 + nc;
  const int nb   = na + 16;
  const int nodeB = blockIdx.x * 16;

  // Stage A = [hF[node] | f16(msg[node])]
  for (int c = tid; c < 16 * 64; c += 256) {
    int m  = c >> 6;
    int k8 = (c & 63) * 8;
    int node = nodeB + m;
    if (k8 < 256) {
      *(v8h*)(sA + m * A_STR + k8) = *(const v8h*)(hF + (size_t)node * HD + k8);
    } else {
      const float* s = msg + (size_t)node * HD + (k8 - 256);
      _Float16* d = sA + m * A_STR + k8;
#pragma unroll
      for (int i = 0; i < 8; ++i) d[i] = (_Float16)s[i];
    }
  }
  __syncthreads();

  // GEMM1: [16x512] @ n_w1 -> silu -> sM
  {
    v8f c0 = {}, c1 = {};
#pragma unroll
    for (int ks = 0; ks < 16; ++ks) {
      v16h a  = afrag_lds(sA, A_STR, ks * 32, lane);
      v16h b0 = bfrag_glb(nW1T, 512, na, ks * 32, lane);
      v16h b1 = bfrag_glb(nW1T, 512, nb, ks * 32, lane);
      c0 = __builtin_amdgcn_wmma_f32_16x16x32_f16(false, a, false, b0, (short)0, c0, false, false);
      c1 = __builtin_amdgcn_wmma_f32_16x16x32_f16(false, a, false, b1, (short)0, c1, false, false);
    }
    float baA = nB1[na], baB = nB1[nb];
#pragma unroll
    for (int r = 0; r < 8; ++r) {
      int m = r + 8 * hi;
      sM[m * M_STR + na] = (_Float16)silu_f(c0[r] + baA);
      sM[m * M_STR + nb] = (_Float16)silu_f(c1[r] + baB);
    }
  }
  __syncthreads();

  // GEMM2: hidden @ n_w2 + b2 + residual h -> sY
  {
    v8f c0 = {}, c1 = {};
#pragma unroll
    for (int ks = 0; ks < 8; ++ks) {
      v16h a  = afrag_lds(sM, M_STR, ks * 32, lane);
      v16h b0 = bfrag_glb(nW2T, 256, na, ks * 32, lane);
      v16h b1 = bfrag_glb(nW2T, 256, nb, ks * 32, lane);
      c0 = __builtin_amdgcn_wmma_f32_16x16x32_f16(false, a, false, b0, (short)0, c0, false, false);
      c1 = __builtin_amdgcn_wmma_f32_16x16x32_f16(false, a, false, b1, (short)0, c1, false, false);
    }
    float baA = nB2[na], baB = nB2[nb];
#pragma unroll
    for (int r = 0; r < 8; ++r) {
      int m = r + 8 * hi;
      size_t rb = (size_t)(nodeB + m) * HD;
      sY[m * Y_STR + na] = c0[r] + baA + h[rb + na];
      sY[m * Y_STR + nb] = c1[r] + baB + h[rb + nb];
    }
  }
  __syncthreads();

  // LayerNorm: 16 lanes per row; each lane covers cols {sub, sub+16, ...}.
  {
    int row = tid >> 4;           // lanes 0-15 of wave w -> row 2w; lanes 16-31 -> row 2w+1
    int sub = tid & 15;
    const float* yr = sY + row * Y_STR;
    float s = 0.0f;
#pragma unroll
    for (int j = 0; j < 16; ++j) s += yr[j * 16 + sub];
    s += __shfl_xor(s, 8, 32); s += __shfl_xor(s, 4, 32);
    s += __shfl_xor(s, 2, 32); s += __shfl_xor(s, 1, 32);
    float mu = s * (1.0f / 256.0f);
    float vv = 0.0f;
#pragma unroll
    for (int j = 0; j < 16; ++j) { float d = yr[j * 16 + sub] - mu; vv += d * d; }
    vv += __shfl_xor(vv, 8, 32); vv += __shfl_xor(vv, 4, 32);
    vv += __shfl_xor(vv, 2, 32); vv += __shfl_xor(vv, 1, 32);
    float rs = rsqrtf(vv * (1.0f / 256.0f) + LN_EPS);
    size_t rb = (size_t)(nodeB + row) * HD;
#pragma unroll
    for (int j = 0; j < 16; ++j) {
      int col = j * 16 + sub;
      hOut[rb + col] = (yr[col] - mu) * rs * lnG[col] + lnB[col];
    }
  }
}

__global__ __launch_bounds__(256) void egnn_xnew(const float* __restrict__ x,
                                                 float* __restrict__ outX, int n3) {
  int i = blockIdx.x * 256 + threadIdx.x;
  if (i < n3) outX[i] = outX[i] + x[i];   // outX currently holds coord_agg
}

extern "C" void kernel_launch(void* const* d_in, const int* in_sizes, int n_in,
                              void* d_out, int out_size, void* d_ws, size_t ws_size,
                              hipStream_t stream) {
  const float* h   = (const float*)d_in[0];
  const float* x   = (const float*)d_in[1];
  const int*   ei  = (const int*)d_in[2];
  const float* eW1 = (const float*)d_in[3];
  const float* eB1 = (const float*)d_in[4];
  const float* eW2 = (const float*)d_in[5];
  const float* eB2 = (const float*)d_in[6];
  const float* cW1 = (const float*)d_in[7];
  const float* cB1 = (const float*)d_in[8];
  const float* cW2 = (const float*)d_in[9];
  const float* nW1 = (const float*)d_in[10];
  const float* nB1 = (const float*)d_in[11];
  const float* nW2 = (const float*)d_in[12];
  const float* nB2 = (const float*)d_in[13];
  const float* lnG = (const float*)d_in[14];
  const float* lnB = (const float*)d_in[15];

  float* out   = (float*)d_out;
  float* msg   = out;                           // [N][256] accumulator -> later h_out (in place)
  float* outX  = out + (size_t)NN * HD;         // [N][3]  accumulator -> later x_new (in place)

  // Workspace layout (f16 halves); all offsets 16B aligned.
  _Float16* ws16 = (_Float16*)d_ws;
  size_t off = 0;
  _Float16* hF   = ws16 + off; off += (size_t)NN * HD;  // 10.24 MB
  _Float16* eW1T = ws16 + off; off += 256 * 512;
  _Float16* eW2T = ws16 + off; off += 256 * 256;
  _Float16* cW1T = ws16 + off; off += 256 * 256;
  _Float16* nW1T = ws16 + off; off += 256 * 512;
  _Float16* nW2T = ws16 + off; off += 256 * 256;
  (void)ws_size; (void)in_sizes; (void)n_in; (void)out_size;

  // Zero both scatter accumulators (graph-capture-safe memset node).
  hipMemsetAsync(d_out, 0, sizeof(float) * ((size_t)NN * HD + (size_t)NN * 3), stream);

  // Precision prep: h -> f16, weights -> f16 transposed [N][K].
  k_f32_to_f16<<<(NN * HD + 255) / 256, 256, 0, stream>>>(h, hF, NN * HD);
  k_transpose_f16<<<(256 * 512 + 255) / 256, 256, 0, stream>>>(eW1, eW1T, 512, 256);
  k_transpose_f16<<<(256 * 256 + 255) / 256, 256, 0, stream>>>(eW2, eW2T, 256, 256);
  k_transpose_f16<<<(256 * 256 + 255) / 256, 256, 0, stream>>>(cW1, cW1T, 256, 256);
  k_transpose_f16<<<(256 * 512 + 255) / 256, 256, 0, stream>>>(nW1, nW1T, 512, 256);
  k_transpose_f16<<<(256 * 256 + 255) / 256, 256, 0, stream>>>(nW2, nW2T, 256, 256);

  // Edge phase: 16 edges per workgroup.
  egnn_edge_kernel<<<NE / 16, 256, 0, stream>>>(
      hF, x, ei, eW1T, eW1 + (size_t)512 * 256, eB1, eW2T, eB2, cW1T, cB1, cW2, msg, outX);

  // Node phase: 16 nodes per workgroup (consumes msg, writes h_out in place).
  egnn_node_kernel<<<NN / 16, 256, 0, stream>>>(
      h, hF, msg, nW1T, nB1, nW2T, nB2, lnG, lnB, msg);

  // x_new = x + coord_agg (in place).
  egnn_xnew<<<(NN * 3 + 255) / 256, 256, 0, stream>>>(x, outX, NN * 3);
}
